// PursuitConvEncGNN_6837587935251
// MI455X (gfx1250) — matrix operand order repaired and checked by reading, not compiled, not run on hardware
//
#include <hip/hip_runtime.h>

typedef __attribute__((ext_vector_type(16))) __bf16 v16bf;
typedef __attribute__((ext_vector_type(8)))  __bf16 v8bf;
typedef __attribute__((ext_vector_type(8)))  float  v8f;

#define NN 147
#define NPIX (147*147)          // 21609
#define BATCH 64
#define TPBATCH 1351            // ceil(21609/16) pixel tiles per batch

// ---------------------------------------------------------------------------
// conv1 as implicit-GEMM WMMA: M = 16 pixels/tile, N = 16 out ch, K = 27 -> 32.
// k = tap*3 + c. All tap offsets are compile-time immediates off one base ptr.
// Writes NHWC bf16 h1 with coalesced 16-channel row stores.
// ---------------------------------------------------------------------------
__global__ __launch_bounds__(256) void k_conv1_wmma(const float* __restrict__ obs,
                                                    const float* __restrict__ w1,
                                                    const float* __restrict__ b1,
                                                    __bf16* __restrict__ h1) {
    const int lane = threadIdx.x & 31;
    const int n    = lane & 15;
    const int hi   = lane >> 4;
    const int wave = blockIdx.x*8 + (threadIdx.x >> 5);
    if (wave >= BATCH*TPBATCH) return;          // wave-uniform

    // Weight fragment: b[e] = B[hi*8 + (e<8 ? e : e+8)][n], B[k][oc] = w1[oc][c][tap], k=tap*3+c
    v16bf wfrag = {};
    #pragma unroll
    for (int e = 0; e < 16; ++e) {
        const int kA = (e < 8) ? e : e + 8;                 // hi == 0
        const int kB = 8 + kA;                              // hi == 1
        const int iA = (kA % 3)*9 + (kA / 3);
        const int iB = (kB < 27) ? (kB % 3)*9 + (kB / 3) : 0;
        float wv = (hi == 0) ? w1[n*27 + iA]
                             : ((kB < 27) ? w1[n*27 + iB] : 0.f);
        wfrag[e] = (__bf16)wv;
    }
    const float bias_n = b1[n];

    const int b    = wave / TPBATCH;
    const int base = (wave % TPBATCH) * 16;
    const int pix  = base + n;                   // A-row m = lane&15
    const bool rv  = pix < NPIX;
    const int  y   = rv ? pix / NN : 0;
    const int  x   = rv ? pix % NN : 0;

    // 9-bit tap validity mask from 4 edge tests (folds at compile time per tap)
    const bool top = y > 0, bot = y < NN-1, lef = x > 0, rig = x < NN-1;
    int vmask = 0;
    #pragma unroll
    for (int tap = 0; tap < 9; ++tap) {
        const int dy = tap/3 - 1, dx = tap%3 - 1;
        bool ok = (dy < 0 ? top : (dy > 0 ? bot : true)) &&
                  (dx < 0 ? lef : (dx > 0 ? rig : true));
        vmask |= (int)ok << tap;
    }
    const float* pbase = obs + ((size_t)b*NPIX + pix)*3;

    // A fragment: a[e] = A[m][k], k = hi*8 + (e<8 ? e : e+8)
    v16bf a = {};
    #pragma unroll
    for (int e = 0; e < 16; ++e) {
        const int k0v  = (e < 8) ? e : e + 8;       // hi == 0
        const int k1v  = 8 + k0v;                   // hi == 1
        const int tap0 = k0v/3,  c0 = k0v%3;
        const int tap1 = k1v/3,  c1 = k1v%3;
        const int off0 = ((tap0/3 - 1)*NN + (tap0%3 - 1))*3 + c0;
        const int off1 = (k1v < 27) ? ((tap1/3 - 1)*NN + (tap1%3 - 1))*3 + c1 : 0;
        int  tap = hi ? tap1 : tap0;
        int  off = hi ? off1 : off0;
        bool kok = hi ? (k1v < 27) : true;          // k0v <= 23 always valid
        bool ok  = rv && kok && ((vmask >> tap) & 1);
        float v = ok ? pbase[off] : 0.f;
        a[e] = (__bf16)v;
    }

    v8f acc = {};
    acc = __builtin_amdgcn_wmma_f32_16x16x32_bf16(false, a, false, wfrag,
                                                  (short)0, acc, false, false);

    // D row m = v + 8*hi; 16 lanes of a group -> 16 contiguous channels of row m
    __bf16* hb = h1 + (((size_t)b*NPIX + base) << 4) + hi*128 + n;
    if (base + 16 <= NPIX) {                     // wave-uniform fast path
        #pragma unroll
        for (int v = 0; v < 8; ++v)
            hb[v*16] = (__bf16)fmaxf(acc[v] + bias_n, 0.f);
    } else {
        #pragma unroll
        for (int v = 0; v < 8; ++v)
            if (base + v + 8*hi < NPIX)
                hb[v*16] = (__bf16)fmaxf(acc[v] + bias_n, 0.f);
    }
}

// ---------------------------------------------------------------------------
// conv2 as implicit GEMM with v_wmma_f32_16x16x32_bf16.
// M = pixels (16/tile), N = 32 out channels (2 tiles), K = 144 padded to 160.
// All 10 tap loads hoisted ahead of the 10-WMMA chain; tap addresses are
// immediate offsets from one per-pixel base pointer.
// Fused epilogue: bias+ReLU, channel-mean -> ew, diagonal -> xdiag.
// ---------------------------------------------------------------------------
__global__ __launch_bounds__(256) void k_conv2_wmma(const __bf16* __restrict__ h1,
                                                    const float* __restrict__ w2,
                                                    const float* __restrict__ b2,
                                                    float* __restrict__ ew,
                                                    float* __restrict__ xdiag) {
    const int lane = threadIdx.x & 31;
    const int n    = lane & 15;
    const int hi   = lane >> 4;
    const int wave = blockIdx.x*8 + (threadIdx.x >> 5);

    // Weight fragments in registers for the whole kernel.
    v16bf bw[5][2];
    #pragma unroll
    for (int k0 = 0; k0 < 5; ++k0)
        #pragma unroll
        for (int nt = 0; nt < 2; ++nt) {
            v16bf f = {};
            int oc = nt*16 + n;
            #pragma unroll
            for (int e = 0; e < 16; ++e) {
                int tap = 2*k0 + (e >> 3);
                if (tap < 9)
                    f[e] = (__bf16)w2[oc*144 + (hi*8 + (e & 7))*9 + tap];   // OIHW [32][16][3][3]
            }
            bw[k0][nt] = f;
        }
    const float bias0 = b2[n], bias1 = b2[16 + n];

    for (int it = 0; it < 4; ++it) {
        int wt = wave*4 + it;
        if (wt >= BATCH*TPBATCH) break;            // wave-uniform
        const int b    = wt / TPBATCH;
        const int base = (wt % TPBATCH) * 16;

        const int  pix = base + n;                 // A-row m = lane&15
        const bool rv  = pix < NPIX;
        const int  y   = rv ? pix / NN : 0;
        const int  x   = rv ? pix % NN : 0;

        const bool top = y > 0, bot = y < NN-1, lef = x > 0, rig = x < NN-1;
        int vmask = 0;
        #pragma unroll
        for (int tap = 0; tap < 9; ++tap) {
            const int dy = tap/3 - 1, dx = tap%3 - 1;
            bool ok = (dy < 0 ? top : (dy > 0 ? bot : true)) &&
                      (dx < 0 ? lef : (dx > 0 ? rig : true));
            vmask |= (int)ok << tap;
        }
        const __bf16* pbase = h1 + (((size_t)b*NPIX + pix) << 4) + hi*8;

        // Hoist all 10 tap loads (tap 9 = K-padding = zeros)
        v8bf rr[10];
        v8bf zz = {};
        rr[9] = zz;
        #pragma unroll
        for (int tap = 0; tap < 9; ++tap) {
            const int off = ((tap/3 - 1)*NN + (tap%3 - 1)) << 4;   // immediate
            v8bf z = {};
            if (rv && ((vmask >> tap) & 1))
                z = *(const v8bf*)(pbase + off);
            rr[tap] = z;
        }

        // 10 back-to-back WMMAs
        v8f acc0 = {}, acc1 = {};
        #pragma unroll
        for (int k0 = 0; k0 < 5; ++k0) {
            v16bf a;
            #pragma unroll
            for (int e = 0; e < 8; ++e) { a[e] = rr[2*k0][e]; a[8+e] = rr[2*k0+1][e]; }
            acc0 = __builtin_amdgcn_wmma_f32_16x16x32_bf16(false, a, false, bw[k0][0],
                                                           (short)0, acc0, false, false);
            acc1 = __builtin_amdgcn_wmma_f32_16x16x32_bf16(false, a, false, bw[k0][1],
                                                           (short)0, acc1, false, false);
        }
        #pragma unroll
        for (int v = 0; v < 8; ++v) {
            acc0[v] = fmaxf(acc0[v] + bias0, 0.f);
            acc1[v] = fmaxf(acc1[v] + bias1, 0.f);
        }

        // ew: mean over 32 channels; row m = v + 8*hi lives across a 16-lane group
        const bool fulltile = (base + 16) <= NPIX;           // wave-uniform
        float* ewb = ew + (size_t)b*NPIX + base + 8*hi;
        #pragma unroll
        for (int v = 0; v < 8; ++v) {
            float s = acc0[v] + acc1[v];
            s += __shfl_xor(s, 1, 16);
            s += __shfl_xor(s, 2, 16);
            s += __shfl_xor(s, 4, 16);
            s += __shfl_xor(s, 8, 16);
            if (n == v && (fulltile || base + v + 8*hi < NPIX))
                ewb[v] = s * (1.f/32.f);
        }

        // diagonal node features: diag pixels are p = 148*i, spacing 148 > 16,
        // so each tile holds at most one diagonal row.
        {
            int rem = base % 148;
            int off = (rem == 0) ? 0 : 148 - rem;            // wave-uniform
            if (off < 16 && base + off < NPIX) {
                int yy = (base + off) / 148;
                if (hi == (off >> 3)) {
                    float d0 = 0.f, d1 = 0.f;
                    #pragma unroll
                    for (int v = 0; v < 8; ++v)
                        if ((off & 7) == v) { d0 = acc0[v]; d1 = acc1[v]; }
                    float* d = xdiag + ((size_t)b*NN + yy)*32;
                    d[n]      = d0;
                    d[16 + n] = d1;
                }
            }
        }
    }
}

// ---------------------------------------------------------------------------
// degree / D^-1/2  (deg[t] = 1 + sum_s ew[b,s,t], > 0 due to self loop)
// ---------------------------------------------------------------------------
__global__ void k_deg(const float* __restrict__ ew, float* __restrict__ dinv) {
    int b = blockIdx.x, t = threadIdx.x;    // 160 threads
    float v = 0.f;
    if (t < NN) {
        float s = 1.f;
        for (int ss = 0; ss < NN; ++ss)
            s += ew[(size_t)b*NPIX + ss*NN + t];
        v = rsqrtf(s);
    }
    dinv[b*160 + t] = v;
}

// norm[b,t,s] = dinv[t]*(ew[b,s,t] + (s==t))*dinv[s], zero-padded to 160x160 bf16
__global__ void k_norm(const float* __restrict__ ew, const float* __restrict__ dinv,
                       __bf16* __restrict__ nrm) {
    size_t i = (size_t)blockIdx.x*256 + threadIdx.x;   // BATCH*160*160
    int s = (int)(i % 160);
    size_t r = i / 160;
    int t = (int)(r % 160);
    int b = (int)(r / 160);
    float v = 0.f;
    if (t < NN && s < NN) {
        float a = ew[(size_t)b*NPIX + s*NN + t] + (s == t ? 1.f : 0.f);
        v = dinv[b*160 + t] * a * dinv[b*160 + s];
    }
    nrm[i] = (__bf16)v;
}

// ---------------------------------------------------------------------------
// feat: xwT[b,f,s] = bf16( sum_k x[b,s,k] * W[k,f] ), s padded 147->160 w/ zeros.
// Transposed layout makes WMMA B-fragments contiguous 16B loads.
// ---------------------------------------------------------------------------
template<int K, int F>
__global__ void k_feat(const float* __restrict__ x, const float* __restrict__ W,
                       __bf16* __restrict__ xwT) {
    int bs = blockIdx.x;            // b*160 + s
    int b = bs / 160, s = bs % 160;
    int f = threadIdx.x;            // F threads
    __shared__ float xs[K];
    float val = 0.f;
    if (s < NN) {
        for (int k = f; k < K; k += F) xs[k] = x[((size_t)b*NN + s)*K + k];
        __syncthreads();
        float acc = 0.f;
        #pragma unroll 8
        for (int k = 0; k < K; ++k) acc += xs[k] * W[k*F + f];
        val = acc;
    }
    xwT[((size_t)b*F + f)*160 + s] = (__bf16)val;
}

// ---------------------------------------------------------------------------
// agg: y[b,t,f] = relu(bias[f] + sum_s norm[b,t,s] * xw[b,s,f]) via bf16 WMMA.
// One wave per (b, m-tile, n-tile); K = 160 -> 5 WMMA steps; all 20 fragment
// loads hoisted ahead of the WMMA chain; immediate-offset row stores.
// ---------------------------------------------------------------------------
template<int F>
__global__ __launch_bounds__(256) void k_agg(const __bf16* __restrict__ nrm,
                                             const __bf16* __restrict__ xwT,
                                             const float* __restrict__ bias,
                                             float* __restrict__ yout) {
    const int NT = F/16;
    int lane = threadIdx.x & 31;
    int wave = blockIdx.x*8 + (threadIdx.x >> 5);
    if (wave >= BATCH*10*NT) return;            // wave-uniform
    int nt = wave % NT;
    int mt = (wave / NT) % 10;
    int b  = wave / (NT*10);
    int n = lane & 15, hi = lane >> 4;

    const __bf16* arow = nrm + ((size_t)b*160 + mt*16 + n)*160;   // row t, K-contig
    const __bf16* brow = xwT + ((size_t)b*F   + nt*16 + n)*160;   // col f, K-contig

    v8bf A0[5], A1[5], B0[5], B1[5];
    #pragma unroll
    for (int k0 = 0; k0 < 5; ++k0) {
        A0[k0] = *(const v8bf*)(arow + k0*32 + hi*8);
        A1[k0] = *(const v8bf*)(arow + k0*32 + hi*8 + 16);
        B0[k0] = *(const v8bf*)(brow + k0*32 + hi*8);
        B1[k0] = *(const v8bf*)(brow + k0*32 + hi*8 + 16);
    }
    v8f acc = {};
    #pragma unroll
    for (int k0 = 0; k0 < 5; ++k0) {
        v16bf af, bfm;
        #pragma unroll
        for (int e = 0; e < 8; ++e) {
            af[e]  = A0[k0][e]; af[8+e]  = A1[k0][e];
            bfm[e] = B0[k0][e]; bfm[8+e] = B1[k0][e];
        }
        acc = __builtin_amdgcn_wmma_f32_16x16x32_bf16(false, af, false, bfm,
                                                      (short)0, acc, false, false);
    }
    float bv = bias[nt*16 + n];
    // D row t = mt*16 + v + 8*hi; base pointer + immediate v*F offsets
    float* yb = yout + ((size_t)b*NN + mt*16 + 8*hi)*F + nt*16 + n;
    if (mt < 9) {                                // wave-uniform: all 16 rows valid
        #pragma unroll
        for (int v = 0; v < 8; ++v)
            yb[v*F] = fmaxf(acc[v] + bv, 0.f);
    } else {
        #pragma unroll
        for (int v = 0; v < 8; ++v)
            if (144 + v + 8*hi < NN)
                yb[v*F] = fmaxf(acc[v] + bv, 0.f);
    }
}

// ---------------------------------------------------------------------------
// global mean pool + policy head
// ---------------------------------------------------------------------------
__global__ void k_pool_policy(const float* __restrict__ y, const float* __restrict__ pw,
                              const float* __restrict__ pb, float* __restrict__ out) {
    int b = blockIdx.x, f = threadIdx.x;     // 128 threads
    __shared__ float sp[128];
    float s = 0.f;
    for (int nn = 0; nn < NN; ++nn) s += y[((size_t)b*NN + nn)*128 + f];
    sp[f] = s * (1.f/147.f);
    __syncthreads();
    if (f < 5) {
        float o = pb[f];
        #pragma unroll 8
        for (int k = 0; k < 128; ++k) o += sp[k] * pw[k*5 + f];
        out[b*5 + f] = o;
    }
}

// ---------------------------------------------------------------------------
extern "C" void kernel_launch(void* const* d_in, const int* in_sizes, int n_in,
                              void* d_out, int out_size, void* d_ws, size_t ws_size,
                              hipStream_t stream) {
    const float* obs = (const float*)d_in[0];
    const float* w1  = (const float*)d_in[1];
    const float* b1  = (const float*)d_in[2];
    const float* w2  = (const float*)d_in[3];
    const float* b2  = (const float*)d_in[4];
    const float* g1w = (const float*)d_in[5];
    const float* g1b = (const float*)d_in[6];
    const float* g2w = (const float*)d_in[7];
    const float* g2b = (const float*)d_in[8];
    const float* g3w = (const float*)d_in[9];
    const float* g3b = (const float*)d_in[10];
    const float* pw  = (const float*)d_in[11];
    const float* pb  = (const float*)d_in[12];
    float* out = (float*)d_out;

    char* wsp = (char*)d_ws;
    auto alloc = [&](size_t bytes) {
        char* p = wsp;
        wsp += (bytes + 255) & ~(size_t)255;
        return p;
    };
    __bf16* h1    = (__bf16*)alloc((size_t)BATCH*NPIX*16*2);   // 44.3 MB conv1 out, NHWC bf16
    float*  ew    = (float*) alloc((size_t)BATCH*NPIX*4);      //  5.5 MB edge weights
    float*  xdiag = (float*) alloc((size_t)BATCH*NN*32*4);     //  1.2 MB node features
    float*  dinv  = (float*) alloc((size_t)BATCH*160*4);
    __bf16* nrm   = (__bf16*)alloc((size_t)BATCH*160*160*2);   //  3.3 MB padded norm, bf16
    __bf16* xwT   = (__bf16*)alloc((size_t)BATCH*128*160*2);   //  2.6 MB transposed xW, bf16
    float*  ya    = (float*) alloc((size_t)BATCH*NN*128*4);    //  4.8 MB
    float*  yb    = (float*) alloc((size_t)BATCH*NN*128*4);    //  4.8 MB

    k_conv1_wmma<<<(BATCH*TPBATCH + 7)/8, 256, 0, stream>>>(obs, w1, b1, h1);

    int waves2 = (BATCH*TPBATCH + 3)/4;
    k_conv2_wmma<<<(waves2 + 7)/8, 256, 0, stream>>>(h1, w2, b2, ew, xdiag);

    k_deg <<<BATCH, 160, 0, stream>>>(ew, dinv);
    k_norm<<<(BATCH*160*160)/256, 256, 0, stream>>>(ew, dinv, nrm);

    k_feat<32,64>  <<<BATCH*160,  64, 0, stream>>>(xdiag, g1w, xwT);
    k_agg<64>      <<<(BATCH*10*4 + 7)/8, 256, 0, stream>>>(nrm, xwT, g1b, ya);
    k_feat<64,128> <<<BATCH*160, 128, 0, stream>>>(ya, g2w, xwT);
    k_agg<128>     <<<(BATCH*10*8 + 7)/8, 256, 0, stream>>>(nrm, xwT, g2b, yb);
    k_feat<128,128><<<BATCH*160, 128, 0, stream>>>(yb, g3w, xwT);
    k_agg<128>     <<<(BATCH*10*8 + 7)/8, 256, 0, stream>>>(nrm, xwT, g3b, ya);

    k_pool_policy<<<BATCH, 128, 0, stream>>>(ya, pw, pb, out);
}
